// BMVC_8323646619675
// MI455X (gfx1250) — matrix operand order repaired and checked by reading, not compiled, not run on hardware
//
#include <hip/hip_runtime.h>
#include <cstdint>
#include <cstddef>

// ---------------------------------------------------------------------------
// MI455X (gfx1250): bf16 WMMA GEMMs (f32 accumulate) for every matmul,
// TDM (tensor_load_to_lds) row fetch for the KNN top-k, in-place top-k +
// scatter, fused symmetrize/blend for the masks, Cholesky-QR for the two
// 4096x10 orthonormalizations.
// ---------------------------------------------------------------------------

typedef __attribute__((ext_vector_type(16))) __bf16        v16bf;
typedef __attribute__((ext_vector_type(8)))  float         v8f;
typedef __attribute__((ext_vector_type(8)))  unsigned int  v8u;
typedef __attribute__((ext_vector_type(4)))  unsigned int  u32x4;
typedef __attribute__((ext_vector_type(8)))  int           i32x8;
typedef __attribute__((ext_vector_type(4)))  int           i32x4;

union FragAB { v8u u; v16bf bf; };

__device__ __forceinline__ unsigned short f2bf(float f) {
  unsigned int u = __float_as_uint(f);
  u += 0x7fffu + ((u >> 16) & 1u);          // round-to-nearest-even
  return (unsigned short)(u >> 16);
}
__device__ __forceinline__ unsigned int pkbf(float a, float b) {
  return (unsigned int)f2bf(a) | ((unsigned int)f2bf(b) << 16);
}

enum { EPI_NONE = 0, EPI_RELU = 1, EPI_SIGMOID = 2, EPI_SOFTPLUS_CLIP = 3, EPI_EXP_CLIP = 4 };

template <int EPI>
__device__ __forceinline__ float epilogue(float z) {
  if constexpr (EPI == EPI_RELU) {
    return fmaxf(z, 0.f);
  } else if constexpr (EPI == EPI_SIGMOID) {
    return 1.f / (1.f + __expf(-z));
  } else if constexpr (EPI == EPI_SOFTPLUS_CLIP) {
    float sp = (z > 20.f) ? z : log1pf(__expf(z));
    return fminf(fmaxf(sp, 1e-4f), 1e4f);
  } else if constexpr (EPI == EPI_EXP_CLIP) {
    return fminf(fmaxf(__expf(z), 1e-5f), 1e6f);
  } else {
    return z;
  }
}

// ---------------------------------------------------------------------------
// Tiled GEMM: C[M,N] = epi(A[M,K] * B(+T)[K,N] + bias[N])
// 256 threads = 8 wave32 waves in a 4x2 grid; block tile 128x64; each wave
// owns a 32x32 tile = 2x2 WMMAs (2 A-frags + 2 B-frags = 8 ds_load_b128 per
// 4 v_wmma). Interior tiles use vectorized staging (b128 loads, packed b128
// LDS stores); edge tiles fall back to guarded scalar staging.
// ---------------------------------------------------------------------------
template <int EPI, bool TRANSB>
__global__ __launch_bounds__(256) void gemm_wmma_bf16(
    const float* __restrict__ A, const float* __restrict__ B,
    const float* __restrict__ bias, float* __restrict__ C,
    int M, int N, int K, int lda, int ldb, int ldc)
{
  constexpr int LS = 40;                       // padded halfword stride
  __shared__ __align__(16) unsigned short As[128 * LS];  // [m][k]
  __shared__ __align__(16) unsigned short Bs[64 * LS];   // [n][k] (B transposed)

  const int tid   = threadIdx.x;
  const int m0    = blockIdx.y * 128;
  const int n0    = blockIdx.x * 64;
  const int lane  = tid & 31;                  // wave32
  const int wv    = tid >> 5;
  const int hi    = lane >> 4;
  const int lm    = lane & 15;
  const int mBase = (wv >> 1) * 32;            // 0,32,64,96
  const int nBase = (wv & 1) * 32;             // 0,32

  const bool fullA  = (m0 + 128 <= M) && ((lda & 3) == 0);
  const bool fullB  = (n0 + 64 <= N);
  const bool fullBT = fullB && ((ldb & 3) == 0);

  v8f c00 = {}, c01 = {}, c10 = {}, c11 = {};

  for (int kt = 0; kt < K; kt += 32) {
    const bool fullK = (kt + 32 <= K);

    // ---------------- stage A tile (128x32) ----------------
    if (fullA && fullK) {
      const int m  = tid >> 1;               // 0..127
      const int ks = (tid & 1) * 16;         // 0,16
      const float* p = A + (size_t)(m0 + m) * lda + (kt + ks);
      const float4 u0 = *(const float4*)p;
      const float4 u1 = *(const float4*)(p + 4);
      const float4 u2 = *(const float4*)(p + 8);
      const float4 u3 = *(const float4*)(p + 12);
      uint4 s0, s1;
      s0.x = pkbf(u0.x, u0.y); s0.y = pkbf(u0.z, u0.w);
      s0.z = pkbf(u1.x, u1.y); s0.w = pkbf(u1.z, u1.w);
      s1.x = pkbf(u2.x, u2.y); s1.y = pkbf(u2.z, u2.w);
      s1.z = pkbf(u3.x, u3.y); s1.w = pkbf(u3.z, u3.w);
      *(uint4*)&As[m * LS + ks]     = s0;
      *(uint4*)&As[m * LS + ks + 8] = s1;
    } else {
      for (int i = tid; i < 128 * 32; i += 256) {
        int m = i >> 5, k = i & 31;
        int gm = m0 + m, gk = kt + k;
        float v = (gm < M && gk < K) ? A[(size_t)gm * lda + gk] : 0.f;
        As[m * LS + k] = f2bf(v);
      }
    }
    // ---------------- stage B tile -> Bs[n][k] ----------------
    if (TRANSB ? (fullBT && fullK) : (fullB && fullK)) {
      if constexpr (TRANSB) {
        const int n  = tid >> 2;             // 0..63
        const int ks = (tid & 3) * 8;
        const float* p = B + (size_t)(n0 + n) * ldb + (kt + ks);
        const float4 u = *(const float4*)p;
        const float4 w = *(const float4*)(p + 4);
        uint4 st;
        st.x = pkbf(u.x, u.y); st.y = pkbf(u.z, u.w);
        st.z = pkbf(w.x, w.y); st.w = pkbf(w.z, w.w);
        *(uint4*)&Bs[n * LS + ks] = st;
      } else {
        const int n  = tid & 63;             // coalesced over n per k
        const int kg = (tid >> 6) * 8;       // 0,8,16,24
        float v[8];
#pragma unroll
        for (int j = 0; j < 8; ++j) v[j] = B[(size_t)(kt + kg + j) * ldb + (n0 + n)];
        uint4 st;
        st.x = pkbf(v[0], v[1]); st.y = pkbf(v[2], v[3]);
        st.z = pkbf(v[4], v[5]); st.w = pkbf(v[6], v[7]);
        *(uint4*)&Bs[n * LS + kg] = st;
      }
    } else {
      for (int i = tid; i < 32 * 64; i += 256) {
        int k = i >> 6, n = i & 63;
        int gk = kt + k, gn = n0 + n;
        float v = 0.f;
        if (gk < K && gn < N)
          v = TRANSB ? B[(size_t)gn * ldb + gk] : B[(size_t)gk * ldb + gn];
        Bs[n * LS + k] = f2bf(v);
      }
    }
    // Prefetch next A tile row (global_prefetch_b8 path).
    if (kt + 32 < K) {
      int pm = m0 + (tid >> 1);
      if (pm < M) __builtin_prefetch(&A[(size_t)pm * lda + (kt + 32)], 0, 1);
    }
    __syncthreads();

    FragAB a0, a1, b0, b1;
#pragma unroll
    for (int v = 0; v < 8; ++v) {
      // A-matrix 16x32 bf16 layout (ISA 7.12.2): K = 16*(v>=4) + 8*hi + 2*(v&3)
      int ka = ((v & 4) ? 16 : 0) + hi * 8 + (v & 3) * 2;
      // B-matrix 32x16 bf16 layout: same K order as A
      int kb = ((v & 4) ? 16 : 0) + hi * 8 + (v & 3) * 2;
      a0.u[v] = *(const unsigned int*)&As[(mBase + lm) * LS + ka];
      a1.u[v] = *(const unsigned int*)&As[(mBase + 16 + lm) * LS + ka];
      b0.u[v] = *(const unsigned int*)&Bs[(nBase + lm) * LS + kb];
      b1.u[v] = *(const unsigned int*)&Bs[(nBase + 16 + lm) * LS + kb];
    }
    c00 = __builtin_amdgcn_wmma_f32_16x16x32_bf16(false, a0.bf, false, b0.bf, (short)0, c00, false, false);
    c01 = __builtin_amdgcn_wmma_f32_16x16x32_bf16(false, a0.bf, false, b1.bf, (short)0, c01, false, false);
    c10 = __builtin_amdgcn_wmma_f32_16x16x32_bf16(false, a1.bf, false, b0.bf, (short)0, c10, false, false);
    c11 = __builtin_amdgcn_wmma_f32_16x16x32_bf16(false, a1.bf, false, b1.bf, (short)0, c11, false, false);
    __syncthreads();
  }

  // Epilogue + store. C/D layout: VGPR r -> row (8*hi + r), col = lane&15.
  const int gn0 = n0 + nBase + lm;
  const int gn1 = gn0 + 16;
  const float bb0 = (bias && gn0 < N) ? bias[gn0] : 0.f;
  const float bb1 = (bias && gn1 < N) ? bias[gn1] : 0.f;
#pragma unroll
  for (int r = 0; r < 8; ++r) {
    int gm0 = m0 + mBase + hi * 8 + r;
    int gm1 = gm0 + 16;
    if (gm0 < M) {
      if (gn0 < N) C[(size_t)gm0 * ldc + gn0] = epilogue<EPI>(c00[r] + bb0);
      if (gn1 < N) C[(size_t)gm0 * ldc + gn1] = epilogue<EPI>(c01[r] + bb1);
    }
    if (gm1 < M) {
      if (gn0 < N) C[(size_t)gm1 * ldc + gn0] = epilogue<EPI>(c10[r] + bb0);
      if (gn1 < N) C[(size_t)gm1 * ldc + gn1] = epilogue<EPI>(c11[r] + bb1);
    }
  }
}

// ---------------------------------------------------------------------------
// TDM: DMA one contiguous row of `nelems` f32 from global into LDS.
// D# layout per CDNA5 ISA 8.3/8.4 (count=1, type=2, data_size=4B, 1-row tile).
// ---------------------------------------------------------------------------
#if defined(__has_builtin)
#if __has_builtin(__builtin_amdgcn_tensor_load_to_lds) && __has_builtin(__builtin_amdgcn_s_wait_tensorcnt)
#define HAVE_TDM 1
#endif
#endif

#define CDNA5_STR2(x) #x
#define CDNA5_STR(x) CDNA5_STR2(x)
#pragma message("CDNA5 probe: __clang_major__ = " CDNA5_STR(__clang_major__))
#if defined(HAVE_TDM)
#pragma message("CDNA5 probe: TDM builtin AVAILABLE -> tensor_load_to_lds path compiled in")
#else
#pragma message("CDNA5 probe: TDM builtin NOT available -> scalar fallback path")
#endif

#if defined(HAVE_TDM)
__device__ __forceinline__ void tdm_load_row_f32(unsigned int lds_addr,
                                                 const float* gsrc,
                                                 unsigned int nelems) {
  const unsigned long long ga = (unsigned long long)(uintptr_t)gsrc;
  u32x4 g0;
  g0[0] = 1u;                                            // count=1 (valid user D#)
  g0[1] = lds_addr;                                      // LDS byte address
  g0[2] = (unsigned int)ga;                              // global_addr[31:0]
  g0[3] = (unsigned int)((ga >> 32) & 0x01FFFFFFull)     // global_addr[56:32]
        | (2u << 30);                                    // type=2 ("image")
  i32x8 g1;
  g1[0] = (int)(2u << 16);                               // data_size=2 -> 4 bytes
  g1[1] = (int)((nelems & 0xFFFFu) << 16);               // tensor_dim0[15:0]
  g1[2] = (int)((nelems >> 16) | (1u << 16));            // tensor_dim0[31:16] | tensor_dim1=1
  g1[3] = (int)((nelems & 0xFFFFu) << 16);               // tile_dim0
  g1[4] = 1;                                             // tile_dim1 = 1
  g1[5] = (int)nelems;                                   // tensor_dim0_stride[31:0]
  g1[6] = 0;
  g1[7] = 0;
  i32x4 gz = {0, 0, 0, 0};
#if __clang_major__ >= 23
  i32x8 gz8 = {0, 0, 0, 0, 0, 0, 0, 0};
  __builtin_amdgcn_tensor_load_to_lds(g0, g1, gz, gz, gz8, 0);
#else
  __builtin_amdgcn_tensor_load_to_lds(g0, g1, gz, gz, 0);
#endif
}
#endif

// ---------------------------------------------------------------------------
// Row squared norms (block per row).
// ---------------------------------------------------------------------------
__global__ __launch_bounds__(256) void rowSqNorm(const float* __restrict__ X,
                                                 float* __restrict__ sq, int D) {
  __shared__ float red[256];
  const int row = blockIdx.x, tid = threadIdx.x;
  float a = 0.f;
  for (int j = tid; j < D; j += 256) { float v = X[(size_t)row * D + j]; a += v * v; }
  red[tid] = a; __syncthreads();
  for (int s = 128; s > 0; s >>= 1) { if (tid < s) red[tid] += red[tid + s]; __syncthreads(); }
  if (tid == 0) sq[row] = red[0];
}

// ---------------------------------------------------------------------------
// In-place KNN: Gram row -> LDS (TDM DMA when available) -> d2 -> zero row ->
// select 26 smallest (first = self, dropped like idx[:,1:]) -> scatter 25
// weights + diag 1.0 back into the same row.
// ---------------------------------------------------------------------------
__global__ __launch_bounds__(256) void knn_topk_inplace(float* __restrict__ G,
                                                        const float* __restrict__ sq,
                                                        int n) {
  const int row = blockIdx.x, tid = threadIdx.x;
  __shared__ float drow[4096];
  __shared__ float sval[256];
  __shared__ int   sidx[256];
  __shared__ float seld[26];
  __shared__ int   seli[26];

  const float  sqi  = sq[row];
  const size_t base = (size_t)row * n;

#if defined(HAVE_TDM)
  if (tid < 32) {                               // wave 0 issues the row DMA
    tdm_load_row_f32((unsigned int)(uintptr_t)(void*)drow, G + base, (unsigned int)n);
    __builtin_amdgcn_s_wait_tensorcnt(0);
  }
  __syncthreads();
  for (int j = tid; j < n; j += 256) drow[j] = sqi + sq[j] - 2.f * drow[j];
#else
  for (int j = tid; j < n; j += 256) drow[j] = sqi + sq[j] - 2.f * G[base + j];
#endif
  __syncthreads();
  for (int j = tid; j < n; j += 256) G[base + j] = 0.f;

  for (int it = 0; it < 26; ++it) {
    float best = 3.4e38f; int bi = n;
    for (int j = tid; j < n; j += 256) {
      float v = drow[j];
      if (v < best) { best = v; bi = j; }
    }
    sval[tid] = best; sidx[tid] = bi; __syncthreads();
    for (int s = 128; s > 0; s >>= 1) {
      if (tid < s) {
        float ov = sval[tid + s]; int oi = sidx[tid + s];
        if (ov < sval[tid] || (ov == sval[tid] && oi < sidx[tid])) { sval[tid] = ov; sidx[tid] = oi; }
      }
      __syncthreads();
    }
    if (tid == 0) { seld[it] = sval[0]; seli[it] = sidx[0]; drow[sidx[0]] = 3.4e38f; }
    __syncthreads();
  }

  if (tid == 0) {
    float dk = seld[25], sum = 0.f;
    for (int j = 1; j < 26; ++j) sum += seld[j];
    float den = 25.f * dk - sum + 1e-8f;
    for (int j = 1; j < 26; ++j) G[base + seli[j]] = (dk - seld[j]) / den;
    G[base + row] = 1.0f;
  }
}

// ---------------------------------------------------------------------------
// mask[i,j] = mask[j,i] = 0.25*(Az_ij + Az_ji + Araw_ij + Araw_ji)
// ---------------------------------------------------------------------------
__global__ void combineMasks(float* __restrict__ Mraw, const float* __restrict__ Az, int n) {
  if (blockIdx.x < blockIdx.y) return;          // upper-triangular blocks only
  const int j = blockIdx.x * 16 + threadIdx.x;
  const int i = blockIdx.y * 16 + threadIdx.y;
  if (i >= n || j >= n || j < i) return;
  const size_t ij = (size_t)i * n + j, ji = (size_t)j * n + i;
  const float v = 0.25f * (Mraw[ij] + Mraw[ji] + Az[ij] + Az[ji]);
  Mraw[ij] = v; Mraw[ji] = v;
}

// ---------------------------------------------------------------------------
// Cholesky-QR support for 4096x10 factors.
// ---------------------------------------------------------------------------
__global__ __launch_bounds__(256) void gram10(const float* __restrict__ Mt,
                                              float* __restrict__ G10, int rows) {
  __shared__ float red[256];
  const int tid = threadIdx.x;
  float acc[100];
#pragma unroll
  for (int i = 0; i < 100; ++i) acc[i] = 0.f;
  for (int r = tid; r < rows; r += 256) {
    float v[10];
#pragma unroll
    for (int c = 0; c < 10; ++c) v[c] = Mt[(size_t)r * 10 + c];
    int e = 0;
#pragma unroll
    for (int a2 = 0; a2 < 10; ++a2)
#pragma unroll
      for (int b2 = 0; b2 < 10; ++b2) acc[e++] += v[a2] * v[b2];
  }
  for (int e = 0; e < 100; ++e) {                 // deterministic tree reduce
    red[tid] = acc[e]; __syncthreads();
    for (int s = 128; s > 0; s >>= 1) { if (tid < s) red[tid] += red[tid + s]; __syncthreads(); }
    if (tid == 0) G10[e] = red[0];
    __syncthreads();
  }
}

__global__ void cholqr10(const float* __restrict__ G, float* __restrict__ Rinv) {
  if (threadIdx.x != 0) return;
  float Lm[10][10];
  for (int i = 0; i < 10; ++i) for (int j = 0; j < 10; ++j) Lm[i][j] = 0.f;
  for (int j = 0; j < 10; ++j) {
    float d = G[j * 10 + j];
    for (int k = 0; k < j; ++k) d -= Lm[j][k] * Lm[j][k];
    d = sqrtf(fmaxf(d, 1e-20f));
    Lm[j][j] = d;
    const float inv = 1.f / d;
    for (int i = j + 1; i < 10; ++i) {
      float s = G[i * 10 + j];
      for (int k = 0; k < j; ++k) s -= Lm[i][k] * Lm[j][k];
      Lm[i][j] = s * inv;
    }
  }
  float Ri[10][10];
  for (int i = 0; i < 10; ++i) for (int j = 0; j < 10; ++j) Ri[i][j] = 0.f;
  for (int c = 0; c < 10; ++c) {
    for (int i = c; i >= 0; --i) {
      float s = (i == c) ? 1.f : 0.f;
      for (int k = i + 1; k <= c; ++k) s -= Lm[k][i] * Ri[k][c];
      Ri[i][c] = s / Lm[i][i];
    }
  }
  for (int i = 0; i < 10; ++i) for (int j = 0; j < 10; ++j) Rinv[i * 10 + j] = Ri[i][j];
}

__global__ void applyRinv10(const float* __restrict__ Mt, const float* __restrict__ Rinv,
                            float* __restrict__ Q, int rows) {
  const int r = blockIdx.x * blockDim.x + threadIdx.x;
  if (r >= rows) return;
  float v[10];
#pragma unroll
  for (int c = 0; c < 10; ++c) v[c] = Mt[(size_t)r * 10 + c];
#pragma unroll
  for (int c = 0; c < 10; ++c) {
    float s = 0.f;
    for (int k = 0; k <= c; ++k) s += v[k] * Rinv[k * 10 + c];
    Q[(size_t)r * 10 + c] = s;
  }
}

// ---------------------------------------------------------------------------
// BatchNorm stats + apply.
// ---------------------------------------------------------------------------
__global__ __launch_bounds__(256) void colStats(const float* __restrict__ H,
                                                float* __restrict__ stats,
                                                int rows, int cols) {
  __shared__ float r1[256], r2[256];
  const int c = blockIdx.x, tid = threadIdx.x;
  float a = 0.f, b = 0.f;
  for (int r = tid; r < rows; r += 256) { float v = H[(size_t)r * cols + c]; a += v; b += v * v; }
  r1[tid] = a; r2[tid] = b; __syncthreads();
  for (int s = 128; s > 0; s >>= 1) {
    if (tid < s) { r1[tid] += r1[tid + s]; r2[tid] += r2[tid + s]; }
    __syncthreads();
  }
  if (tid == 0) {
    float mu = r1[0] / rows;
    stats[c] = mu;
    stats[cols + c] = r2[0] / rows - mu * mu;
  }
}

__global__ void bnRelu(float* __restrict__ H, const float* __restrict__ stats,
                       const float* __restrict__ g, const float* __restrict__ b,
                       int rows, int cols) {
  const int idx = blockIdx.x * blockDim.x + threadIdx.x;
  if (idx >= rows * cols) return;
  const int c = idx % cols;
  const float v = (H[idx] - stats[c]) * rsqrtf(stats[cols + c] + 1e-5f) * g[c] + b[c];
  H[idx] = fmaxf(v, 0.f);
}

// ---------------------------------------------------------------------------
// Host orchestration.
// ---------------------------------------------------------------------------
extern "C" void kernel_launch(void* const* d_in, const int* in_sizes, int n_in,
                              void* d_out, int out_size, void* d_ws, size_t ws_size,
                              hipStream_t stream)
{
  (void)in_sizes; (void)n_in; (void)out_size; (void)ws_size;
  constexpr int Nn = 4096, F = 1000, H1 = 256, H2 = 128, Cc = 10, E = 192;

  const float* x    = (const float*)d_in[0];
  const float* sadj = (const float*)d_in[1];
  const float* fadj = (const float*)d_in[2];
  const float* s_x  = (const float*)d_in[3];
  const float* sW1 = (const float*)d_in[4];  const float* sb1 = (const float*)d_in[5];
  const float* sW2 = (const float*)d_in[6];  const float* sb2 = (const float*)d_in[7];
  const float* fW1 = (const float*)d_in[8];  const float* fb1 = (const float*)d_in[9];
  const float* fW2 = (const float*)d_in[10]; const float* fb2 = (const float*)d_in[11];
  const float* cW1 = (const float*)d_in[12]; const float* cb1 = (const float*)d_in[13];
  const float* cW2 = (const float*)d_in[14]; const float* cb2 = (const float*)d_in[15];
  const float* fuW = (const float*)d_in[16]; const float* fub = (const float*)d_in[17];
  const float* dW  = (const float*)d_in[18]; const float* db  = (const float*)d_in[19];
  const float* bng = (const float*)d_in[20]; const float* bnb = (const float*)d_in[21];
  const float* piW = (const float*)d_in[22]; const float* pib = (const float*)d_in[23];
  const float* dsW = (const float*)d_in[24]; const float* dsb = (const float*)d_in[25];
  const float* mnW = (const float*)d_in[26]; const float* mnb = (const float*)d_in[27];

  float* out   = (float*)d_out;
  float* y1    = out;
  float* y2    = y1    + (size_t)Nn * Cc;
  float* embO  = y2    + (size_t)Nn * Cc;
  float* piO   = embO  + (size_t)Nn * E;
  float* dispO = piO   + (size_t)Nn * F;
  float* meanO = dispO + (size_t)Nn * F;
  float* mask0 = meanO + (size_t)Nn * F;
  float* mask1 = mask0 + (size_t)Nn * Nn;

  float* ws = (float*)d_ws;              // ~82 MB used
  size_t o = 0;
  float* wT1  = ws + o; o += (size_t)Nn * H1;
  float* wH   = ws + o; o += (size_t)Nn * H1;
  float* wT3  = ws + o; o += (size_t)Nn * H2;
  float* wCat = ws + o; o += (size_t)Nn * (2 * H2);
  float* wM1  = ws + o; o += (size_t)Nn * Cc;
  float* wM2  = ws + o; o += (size_t)Nn * Cc;
  float* wGa  = ws + o; o += 128;
  float* wGb  = ws + o; o += 128;
  float* wRa  = ws + o; o += 128;
  float* wRb  = ws + o; o += 128;
  float* wHd  = ws + o; o += (size_t)Nn * H1;
  float* wSt  = ws + o; o += 2 * H1;
  float* wSq  = ws + o; o += Nn;
  float* wG   = ws + o; o += (size_t)Nn * Nn;   // Gram -> in-place kz adjacency

  const dim3 B256(256);
  auto g = [](int M, int N) { return dim3((unsigned)((N + 63) / 64), (unsigned)((M + 127) / 128)); };

  // ---- GCN view s: h = relu(sadj@(x@W1)+b1); emb1 = relu(sadj@(h@W2)+b2)
  gemm_wmma_bf16<EPI_NONE, false><<<g(Nn, H1), B256, 0, stream>>>(x,    sW1, nullptr, wT1,  Nn, H1, F,  F,  H1, H1);
  gemm_wmma_bf16<EPI_RELU, false><<<g(Nn, H1), B256, 0, stream>>>(sadj, wT1, sb1,     wH,   Nn, H1, Nn, Nn, H1, H1);
  gemm_wmma_bf16<EPI_NONE, false><<<g(Nn, H2), B256, 0, stream>>>(wH,   sW2, nullptr, wT3,  Nn, H2, H1, H1, H2, H2);
  gemm_wmma_bf16<EPI_RELU, false><<<g(Nn, H2), B256, 0, stream>>>(sadj, wT3, sb2,     wCat, Nn, H2, Nn, Nn, H2, 2 * H2);
  // ---- GCN view f (emb2 into columns 128..255 of wCat = concat buffer)
  gemm_wmma_bf16<EPI_NONE, false><<<g(Nn, H1), B256, 0, stream>>>(x,    fW1, nullptr, wT1,       Nn, H1, F,  F,  H1, H1);
  gemm_wmma_bf16<EPI_RELU, false><<<g(Nn, H1), B256, 0, stream>>>(fadj, wT1, fb1,     wH,        Nn, H1, Nn, Nn, H1, H1);
  gemm_wmma_bf16<EPI_NONE, false><<<g(Nn, H2), B256, 0, stream>>>(wH,   fW2, nullptr, wT3,       Nn, H2, H1, H1, H2, H2);
  gemm_wmma_bf16<EPI_RELU, false><<<g(Nn, H2), B256, 0, stream>>>(fadj, wT3, fb2,     wCat + H2, Nn, H2, Nn, Nn, H2, 2 * H2);

  // ---- cluster heads + Cholesky-QR -> y1, y2
  gemm_wmma_bf16<EPI_NONE, false><<<g(Nn, Cc), B256, 0, stream>>>(wCat,      cW1, cb1, wM1, Nn, Cc, H2, 2 * H2, Cc, Cc);
  gemm_wmma_bf16<EPI_NONE, false><<<g(Nn, Cc), B256, 0, stream>>>(wCat + H2, cW2, cb2, wM2, Nn, Cc, H2, 2 * H2, Cc, Cc);
  gram10<<<1, B256, 0, stream>>>(wM1, wGa, Nn);
  gram10<<<1, B256, 0, stream>>>(wM2, wGb, Nn);
  cholqr10<<<1, 32, 0, stream>>>(wGa, wRa);
  cholqr10<<<1, 32, 0, stream>>>(wGb, wRb);
  applyRinv10<<<dim3(16), B256, 0, stream>>>(wM1, wRa, y1, Nn);
  applyRinv10<<<dim3(16), B256, 0, stream>>>(wM2, wRb, y2, Nn);

  // ---- fusion + decoder
  gemm_wmma_bf16<EPI_NONE, false><<<g(Nn, E),  B256, 0, stream>>>(wCat, fuW, fub, embO, Nn, E,  2 * H2, 2 * H2, E,  E);
  gemm_wmma_bf16<EPI_NONE, false><<<g(Nn, H1), B256, 0, stream>>>(embO, dW,  db,  wHd,  Nn, H1, E,      E,      H1, H1);
  colStats<<<dim3(H1), B256, 0, stream>>>(wHd, wSt, Nn, H1);
  bnRelu<<<dim3((Nn * H1 + 255) / 256), B256, 0, stream>>>(wHd, wSt, bng, bnb, Nn, H1);
  gemm_wmma_bf16<EPI_SIGMOID,       false><<<g(Nn, F), B256, 0, stream>>>(wHd, piW, pib, piO,   Nn, F, H1, H1, F, F);
  gemm_wmma_bf16<EPI_SOFTPLUS_CLIP, false><<<g(Nn, F), B256, 0, stream>>>(wHd, dsW, dsb, dispO, Nn, F, H1, H1, F, F);
  gemm_wmma_bf16<EPI_EXP_CLIP,      false><<<g(Nn, F), B256, 0, stream>>>(wHd, mnW, mnb, meanO, Nn, F, H1, H1, F, F);

  // ---- kz = knn_graph(emb): Gram (WMMA, B transposed) -> in-place topk
  rowSqNorm<<<dim3(Nn), B256, 0, stream>>>(embO, wSq, E);
  gemm_wmma_bf16<EPI_NONE, true><<<g(Nn, Nn), B256, 0, stream>>>(embO, embO, nullptr, wG, Nn, Nn, E, E, E, Nn);
  knn_topk_inplace<<<dim3(Nn), B256, 0, stream>>>(wG, wSq, Nn);

  // ---- mask0 = 0.5*kz + 0.5*knn_graph(s_x), built in place in d_out
  rowSqNorm<<<dim3(Nn), B256, 0, stream>>>(s_x, wSq, F);
  gemm_wmma_bf16<EPI_NONE, true><<<g(Nn, Nn), B256, 0, stream>>>(s_x, s_x, nullptr, mask0, Nn, Nn, F, F, F, Nn);
  knn_topk_inplace<<<dim3(Nn), B256, 0, stream>>>(mask0, wSq, Nn);
  combineMasks<<<dim3(256, 256), dim3(16, 16), 0, stream>>>(mask0, wG, Nn);

  // ---- mask1 = 0.5*kz + 0.5*knn_graph(x)
  rowSqNorm<<<dim3(Nn), B256, 0, stream>>>(x, wSq, F);
  gemm_wmma_bf16<EPI_NONE, true><<<g(Nn, Nn), B256, 0, stream>>>(x, x, nullptr, mask1, Nn, Nn, F, F, F, Nn);
  knn_topk_inplace<<<dim3(Nn), B256, 0, stream>>>(mask1, wSq, Nn);
  combineMasks<<<dim3(256, 256), dim3(16, 16), 0, stream>>>(mask1, wG, Nn);
}
